// Llama4TextAttention_1614907703552
// MI455X (gfx1250) — matrix operand order repaired and checked by reading, not compiled, not run on hardware
//
#include <hip/hip_runtime.h>
#include <math.h>

// ---------------------------------------------------------------------------
// Llama4 text attention block for gfx1250 (MI455X), bf16 WMMA + flash attention
//   B=1, S=2048, D=4096, H=32, KV=8, HD=128, N_REP=4
//
// Pipeline (all on `stream`):
//   1) fp32 -> bf16 conversion passes (hidden states, weights; one-time)
//   2) QKV projections: pure-bf16 WMMA GEMM, 64x64 tile per wave
//   3) RoPE + l2norm (head-major bf16), V transpose to (KV,HD,S)
//   4) causal flash attention (online softmax, GQA), bf16 output
//   5) Wo projection straight into d_out (fp32)
// ---------------------------------------------------------------------------

typedef __attribute__((ext_vector_type(16))) __bf16 v16bf;
typedef __attribute__((ext_vector_type(8)))  __bf16 v8bf;
typedef __attribute__((ext_vector_type(8)))  float  v8f;

constexpr int S_   = 2048;
constexpr int D_   = 4096;
constexpr int H_   = 32;
constexpr int KV_  = 8;
constexpr int HD_  = 128;
constexpr int NREP = H_ / KV_;
constexpr float SCALING_ = 0.08838834764831845f;   // 128^-0.5

// ---- helpers --------------------------------------------------------------

__device__ __forceinline__ __bf16 f2bf(float f) {
  unsigned u = __builtin_bit_cast(unsigned, f);
  u += 0x7fffu + ((u >> 16) & 1u);                  // round-to-nearest-even
  return __builtin_bit_cast(__bf16, (unsigned short)(u >> 16));
}

__device__ __forceinline__ v16bf cat8(v8bf lo, v8bf hi) {
  return __builtin_shufflevector(lo, hi, 0,1,2,3,4,5,6,7,8,9,10,11,12,13,14,15);
}

// A fragment (16x32 bf16): lane (r,hi) holds row r, K = {hi*8..+7} ∪ {16+hi*8..+7}
__device__ __forceinline__ v16bf load_a_bf16(const __bf16* base, int ld, int row, int k0, int hi) {
  const __bf16* p = base + (size_t)row * ld + k0 + hi * 8;
  return cat8(*(const v8bf*)p, *(const v8bf*)(p + 16));
}
// B fragment (32x16 bf16): lane (r,hi) holds column r, K = {hi*8..+7} ∪ {16+hi*8..+7}
__device__ __forceinline__ v16bf load_b_bf16(const __bf16* base, int ld, int col, int k0, int hi) {
  const __bf16* p = base + (size_t)col * ld + k0 + hi * 8;
  return cat8(*(const v8bf*)p, *(const v8bf*)(p + 16));
}

// ---- fp32 -> bf16 bulk conversion (vectorized, n multiple of 8) -----------

__global__ __launch_bounds__(256) void cvt_bf16(
    const float* __restrict__ X, __bf16* __restrict__ Y, int n8)
{
  int i = blockIdx.x * 256 + threadIdx.x;
  if (i < n8) {
    v8f x = *(const v8f*)(X + (size_t)i * 8);
    v8bf y;
#pragma unroll
    for (int j = 0; j < 8; ++j) y[j] = f2bf(x[j]);
    *(v8bf*)(Y + (size_t)i * 8) = y;
  }
}

// ---- GEMM: C[M][N] = A[M][K] * W[N][K]^T  (bf16 in, WMMA f32 accumulate) --
// grid = (N/64, M/64), block = 32 (one wave per 64x64 output tile)
// Inner loop: 16 global_load_b128 + 16 v_wmma per K=32 step.

__global__ __launch_bounds__(32) void gemm_wmma(
    const __bf16* __restrict__ A, const __bf16* __restrict__ W,
    float* __restrict__ C, int M, int N, int K)
{
  const int lane = threadIdx.x;
  const int r = lane & 15, hi = lane >> 4;
  const int n0 = blockIdx.x * 64;
  const int m0 = blockIdx.y * 64;

  v8f acc[4][4] = {};                                // [mi][nj], 128 VGPRs
  for (int k0 = 0; k0 < K; k0 += 32) {
    v16bf a[4];
#pragma unroll
    for (int mi = 0; mi < 4; ++mi)
      a[mi] = load_a_bf16(A, K, m0 + mi * 16 + r, k0, hi);
#pragma unroll
    for (int nj = 0; nj < 4; ++nj) {
      v16bf b = load_b_bf16(W, K, n0 + nj * 16 + r, k0, hi);
#pragma unroll
      for (int mi = 0; mi < 4; ++mi)
        acc[mi][nj] = __builtin_amdgcn_wmma_f32_16x16x32_bf16(
            false, a[mi], false, b, (short)0, acc[mi][nj], false, false);
    }
  }
#pragma unroll
  for (int mi = 0; mi < 4; ++mi)
#pragma unroll
    for (int nj = 0; nj < 4; ++nj)
#pragma unroll
      for (int v = 0; v < 8; ++v)
        C[(size_t)(m0 + mi * 16 + v + 8 * hi) * N + n0 + nj * 16 + r] = acc[mi][nj][v];
}

// ---- RoPE (interleaved) + l2norm, fp32 (S,NH,HD) -> bf16 (NH,S,HD) ---------
// grid = S*NH, block = 32 (one wave per (s,head) vector of 128)

__global__ __launch_bounds__(32) void rope_norm(
    const float* __restrict__ X, const float* __restrict__ Cs,
    const float* __restrict__ Sn, __bf16* __restrict__ Out, int NH)
{
  const int lane = threadIdx.x;
  const int hh = blockIdx.x % NH;
  const int s  = blockIdx.x / NH;
  const float* x = X + ((size_t)s * NH + hh) * HD_;

  float xr0 = x[2 * lane],      xi0 = x[2 * lane + 1];
  float xr1 = x[2 * lane + 64], xi1 = x[2 * lane + 65];
  float c0 = Cs[s * 64 + lane],      sn0 = Sn[s * 64 + lane];
  float c1 = Cs[s * 64 + lane + 32], sn1 = Sn[s * 64 + lane + 32];
  float or0 = xr0 * c0 - xi0 * sn0, oi0 = xr0 * sn0 + xi0 * c0;
  float or1 = xr1 * c1 - xi1 * sn1, oi1 = xr1 * sn1 + xi1 * c1;

  float sq = or0 * or0 + oi0 * oi0 + or1 * or1 + oi1 * oi1;
#pragma unroll
  for (int m = 1; m < 32; m <<= 1) sq += __shfl_xor(sq, m, 32);
  float inv = rsqrtf(sq * (1.0f / 128.0f) + 1e-6f);

  __bf16* o = Out + ((size_t)hh * S_ + s) * HD_;
  o[2 * lane]      = f2bf(or0 * inv);  o[2 * lane + 1]  = f2bf(oi0 * inv);
  o[2 * lane + 64] = f2bf(or1 * inv);  o[2 * lane + 65] = f2bf(oi1 * inv);
}

// ---- V: fp32 (S,KV,HD) -> bf16 (KV,HD,S) -----------------------------------

__global__ __launch_bounds__(256) void v_transpose(
    const float* __restrict__ V, __bf16* __restrict__ Vt)
{
  int idx = blockIdx.x * 256 + threadIdx.x;           // S*KV*HD = 2M elements
  int d  = idx % HD_;
  int kv = (idx / HD_) % KV_;
  int s  = idx / (HD_ * KV_);
  Vt[((size_t)kv * HD_ + d) * S_ + s] = f2bf(V[idx]);
}

// ---- Flash attention: one wave per (head, 16-row q tile) -------------------
// Q: bf16 (H,S,HD); K: bf16 (KV,S,HD); Vt: bf16 (KV,HD,S); Out: bf16 (S,H*HD)

__global__ __launch_bounds__(32) void flash_attn(
    const __bf16* __restrict__ Qb, const __bf16* __restrict__ Kb,
    const __bf16* __restrict__ Vt, __bf16* __restrict__ Oa)
{
  const int lane = threadIdx.x;
  const int r = lane & 15, hi = lane >> 4;
  const int qb = blockIdx.x;
  const int h  = blockIdx.y;
  const int q0 = qb * 16;

  const __bf16* Qh = Qb + (size_t)h * S_ * HD_;
  const __bf16* Kh = Kb + (size_t)(h / NREP) * S_ * HD_;
  const __bf16* Vh = Vt + (size_t)(h / NREP) * HD_ * S_;

  v16bf qf[4];
#pragma unroll
  for (int c = 0; c < 4; ++c) qf[c] = load_a_bf16(Qh, HD_, q0 + r, c * 32, hi);

  float rmax[8], rsum[8];
#pragma unroll
  for (int v = 0; v < 8; ++v) { rmax[v] = -1e30f; rsum[v] = 0.0f; }
  v8f o[8] = {};

  __shared__ __bf16 p_lds[16 * 32];

  const int nk = (q0 + 47) >> 5;                      // 32-wide key chunks
  for (int kc = 0; kc < nk; ++kc) {
    const int k0 = kc * 32;

    // scores: S(16x32) = Q(16x128) * K^T — two 16x16 C tiles, 8 WMMAs
    v8f s0 = {}, s1 = {};
#pragma unroll
    for (int c = 0; c < 4; ++c) {
      v16bf b0 = load_b_bf16(Kh, HD_, k0 + r,      c * 32, hi);
      v16bf b1 = load_b_bf16(Kh, HD_, k0 + 16 + r, c * 32, hi);
      s0 = __builtin_amdgcn_wmma_f32_16x16x32_bf16(false, qf[c], false, b0, (short)0, s0, false, false);
      s1 = __builtin_amdgcn_wmma_f32_16x16x32_bf16(false, qf[c], false, b1, (short)0, s1, false, false);
    }

    // causal mask + online softmax (row v+8*hi lives in element v across the 16-lane half)
#pragma unroll
    for (int v = 0; v < 8; ++v) {
      const int qpos = q0 + v + 8 * hi;
      float x0 = (k0 + r      > qpos) ? -1e9f : s0[v] * SCALING_;
      float x1 = (k0 + 16 + r > qpos) ? -1e9f : s1[v] * SCALING_;
      float mv = fmaxf(x0, x1);
#pragma unroll
      for (int m = 1; m < 16; m <<= 1) mv = fmaxf(mv, __shfl_xor(mv, m, 32));
      float nm   = fmaxf(rmax[v], mv);
      float corr = __expf(rmax[v] - nm);
      float p0 = __expf(x0 - nm);
      float p1 = __expf(x1 - nm);
      float ps = p0 + p1;
#pragma unroll
      for (int m = 1; m < 16; m <<= 1) ps += __shfl_xor(ps, m, 32);
      rsum[v] = rsum[v] * corr + ps;
      rmax[v] = nm;
#pragma unroll
      for (int j = 0; j < 8; ++j) o[j][v] *= corr;
      // stash P into LDS to reshape C-layout -> A-fragment layout
      p_lds[(v + 8 * hi) * 32 + r]      = f2bf(p0);
      p_lds[(v + 8 * hi) * 32 + 16 + r] = f2bf(p1);
    }
    __syncthreads();

    // P A-fragment: row r, K pattern {hi*8..} ∪ {16+hi*8..}
    const __bf16* pb = p_lds + r * 32 + hi * 8;
    v16bf pf = cat8(*(const v8bf*)pb, *(const v8bf*)(pb + 16));

    // O(16x128) += P(16x32) * V(32x128): 8 WMMAs, V k-rows contiguous in Vt
#pragma unroll
    for (int j = 0; j < 8; ++j) {
      v16bf vb = cat8(*(const v8bf*)(Vh + (size_t)(j * 16 + r) * S_ + k0 + hi * 8), *(const v8bf*)(Vh + (size_t)(j * 16 + r) * S_ + k0 + hi * 8 + 16));
      o[j] = __builtin_amdgcn_wmma_f32_16x16x32_bf16(false, pf, false, vb, (short)0, o[j], false, false);
    }
    __syncthreads();
  }

#pragma unroll
  for (int v = 0; v < 8; ++v) rsum[v] = 1.0f / rsum[v];
#pragma unroll
  for (int j = 0; j < 8; ++j)
#pragma unroll
    for (int v = 0; v < 8; ++v)
      Oa[(size_t)(q0 + v + 8 * hi) * (H_ * HD_) + h * HD_ + j * 16 + r] =
          f2bf(o[j][v] * rsum[v]);
}

// ---------------------------------------------------------------------------

extern "C" void kernel_launch(void* const* d_in, const int* in_sizes, int n_in,
                              void* d_out, int out_size, void* d_ws, size_t ws_size,
                              hipStream_t stream) {
  (void)in_sizes; (void)n_in; (void)out_size; (void)ws_size;
  const float* hs   = (const float*)d_in[0];   // (1, 2048, 4096)
  const float* wq   = (const float*)d_in[1];   // (4096, 4096)
  const float* wk   = (const float*)d_in[2];   // (1024, 4096)
  const float* wv   = (const float*)d_in[3];   // (1024, 4096)
  const float* wo   = (const float*)d_in[4];   // (4096, 4096)
  const float* fcos = (const float*)d_in[5];   // (1, 2048, 64)
  const float* fsin = (const float*)d_in[6];   // (1, 2048, 64)
  float* out = (float*)d_out;                  // (1, 2048, 4096) fp32

  // workspace layout (~144 MB; every region fully produced before consumed,
  // weight slots reused sequentially on the same stream)
  char* ws = (char*)d_ws;
  __bf16* hs_bf   = (__bf16*)(ws);                        // 16 MB (S,D)
  __bf16* wbig_bf = (__bf16*)(ws + (16u  << 20));         // 32 MB slot: wq, then wo
  __bf16* wsml_bf = (__bf16*)(ws + (48u  << 20));         //  8 MB slot: wk, then wv
  float*  q_f32   = (float*) (ws + (56u  << 20));         // 32 MB (S,H,HD)
  float*  k_f32   = (float*) (ws + (88u  << 20));         //  8 MB (S,KV,HD)
  float*  v_f32   = (float*) (ws + (96u  << 20));         //  8 MB (S,KV,HD)
  __bf16* q_bf    = (__bf16*)(ws + (104u << 20));         // 16 MB (H,S,HD)
  __bf16* k_bf    = (__bf16*)(ws + (120u << 20));         //  4 MB (KV,S,HD)
  __bf16* vt_bf   = (__bf16*)(ws + (124u << 20));         //  4 MB (KV,HD,S)
  __bf16* attn_bf = (__bf16*)(ws + (128u << 20));         // 16 MB (S,H*HD)

  dim3 wv32(32);
  const int nHS = (S_ * D_) / 8;           // 1,048,576 v8 chunks
  const int nWB = (D_ * D_) / 8;           // 2,097,152
  const int nWS = (KV_ * HD_ * D_) / 8;    //   524,288

  // 1) activations + first weight to bf16
  cvt_bf16<<<nHS / 256, 256, 0, stream>>>(hs, hs_bf, nHS);

  // 2) QKV projections (pure bf16 WMMA GEMMs)
  cvt_bf16<<<nWB / 256, 256, 0, stream>>>(wq, wbig_bf, nWB);
  gemm_wmma<<<dim3(D_ / 64, S_ / 64), wv32, 0, stream>>>(hs_bf, wbig_bf, q_f32, S_, D_, D_);
  cvt_bf16<<<nWS / 256, 256, 0, stream>>>(wk, wsml_bf, nWS);
  gemm_wmma<<<dim3((KV_ * HD_) / 64, S_ / 64), wv32, 0, stream>>>(hs_bf, wsml_bf, k_f32, S_, KV_ * HD_, D_);
  cvt_bf16<<<nWS / 256, 256, 0, stream>>>(wv, wsml_bf, nWS);
  gemm_wmma<<<dim3((KV_ * HD_) / 64, S_ / 64), wv32, 0, stream>>>(hs_bf, wsml_bf, v_f32, S_, KV_ * HD_, D_);

  // 3) RoPE + l2norm (head-major bf16), V transpose
  rope_norm<<<S_ * H_,  wv32, 0, stream>>>(q_f32, fcos, fsin, q_bf, H_);
  rope_norm<<<S_ * KV_, wv32, 0, stream>>>(k_f32, fcos, fsin, k_bf, KV_);
  v_transpose<<<(S_ * KV_ * HD_) / 256, 256, 0, stream>>>(v_f32, vt_bf);

  // 4) fused causal attention (GQA, online softmax) -> bf16
  flash_attn<<<dim3(S_ / 16, H_), wv32, 0, stream>>>(q_bf, k_bf, vt_bf, attn_bf);

  // 5) output projection straight into d_out
  cvt_bf16<<<nWB / 256, 256, 0, stream>>>(wo, wbig_bf, nWB);
  gemm_wmma<<<dim3(D_ / 64, S_ / 64), wv32, 0, stream>>>(attn_bf, wbig_bf, out, S_, D_, D_);
}
